// KGramMLPSeqModel_56925496541497
// MI455X (gfx1250) — compile-verified, matrix operand order checked by reading, not run on hardware
//
#include <hip/hip_runtime.h>
#include <hip/hip_bf16.h>
#include <stdint.h>

// ---------------------------------------------------------------------------
// k-gram MLP seq model for gfx1250 (MI455X).
// - bf16x3 split-precision GEMM on v_wmma_f32_16x16x32_bf16 (wave32)
// - Tensor Data Mover (tensor_load_to_lds) double-buffered tile staging
// - weights prepacked once to K-major bf16 hi/lo planes (DMA-able tiles)
// ---------------------------------------------------------------------------

#define KWIN   3
#define VOCAB  50257
#define EMBED  1024
#define SEQ    2048
#define BATCH  2
#define M_ROWS (SEQ * BATCH)   // 4096

#define BM 64
#define BN 128
#define BK 32

typedef __attribute__((ext_vector_type(16))) __bf16        v16bf;
typedef __attribute__((ext_vector_type(8)))  __bf16        v8bf;
typedef __attribute__((ext_vector_type(4)))  __bf16        v4bf;
typedef __attribute__((ext_vector_type(8)))  float         v8f;
typedef __attribute__((ext_vector_type(4)))  unsigned int  v4u;
typedef __attribute__((ext_vector_type(8)))  unsigned int  v8u;

// Low 32 bits of a generic pointer to LDS == LDS byte offset (flat aperture
// tag lives in the high dword only).
__device__ inline unsigned lds_off_u32(const void* p) {
    return (unsigned)(uintptr_t)p;
}

// Loop-invariant part of the Tensor DMA descriptor (D# group1, ISA 8.4):
// data_size=2B, tensor_dim0/1 (dim1 = remaining rows -> OOB rows read zero),
// tile_dim0/1, tensor_dim0_stride.  dim0 = Kdim (>= tile_dim0, no K clip).
__device__ inline v8u tdm_g1_2d_bf16(unsigned tileK, unsigned tileRows,
                                     unsigned dim0, unsigned dim1,
                                     unsigned strideElems) {
    v8u g1;
    g1[0] = (1u << 16);                                        // data_size=2B
    g1[1] = (dim0 & 0xffffu) << 16;                            // dim0 lo16
    g1[2] = ((dim0 >> 16) & 0xffffu) | ((dim1 & 0xffffu) << 16);
    g1[3] = ((dim1 >> 16) & 0xffffu) | ((tileK & 0xffffu) << 16);
    g1[4] = tileRows & 0xffffu;                                // tile_dim1
    g1[5] = strideElems;                                       // dim0_stride
    g1[6] = 0u;
    g1[7] = 0u;
    return g1;
}

// Issue one 2-D TDM tile DMA (D# group0, ISA 8.3): global -> LDS compact tile.
__device__ inline void tdm_issue(unsigned ldsOff, const __bf16* gsrc, v8u g1) {
    unsigned long long ga = (unsigned long long)(uintptr_t)gsrc;
    v4u g0;
    g0[0] = 1u;                                                // count=1, user
    g0[1] = ldsOff;                                            // lds_addr
    g0[2] = (unsigned)ga;                                      // gaddr[31:0]
    g0[3] = (unsigned)((ga >> 32) & 0x01ffffffu) | (2u << 30); // gaddr hi, type=2
    asm volatile("tensor_load_to_lds %0, %1" :: "s"(g0), "s"(g1) : "memory");
}

// Per ISA 7.12.2 (16-bit A 16x32): lanes 0-15 hold K=0..7,16..23, lanes 16-31
// hold K=8..15,24..31 -> two 16B chunks at koff and koff+16 of a [row][BK] plane.
__device__ inline v16bf load_frag16(const __bf16* p) {
    v8bf c0 = *(const v8bf*)(p);
    v8bf c1 = *(const v8bf*)(p + 16);
    v16bf r;
#pragma unroll
    for (int i = 0; i < 8; ++i) { r[i] = c0[i]; r[i + 8] = c1[i]; }
    return r;
}

__device__ inline float silu_f(float v) { return v / (1.0f + __expf(-v)); }

// ---------------------------------------------------------------------------
// C[M,N] = act(A[M,K] @ W[K,N] + bias)
// AMODE: 0 = gather rows from E (f32) via tokens; 1 = bf16 hi/lo planes,
//        manual copy; 2 = bf16 hi/lo planes via TDM DMA (double buffered)
// WMODE: 0 = W is f32 [K][N], convert while staging; 2 = prepacked bf16 hi/lo
//        K-major planes [N][K] via TDM DMA
// OUTPLANES: store result as bf16 hi/lo planes (feeds next GEMM) vs f32.
// ---------------------------------------------------------------------------
template <int AMODE, int WMODE, bool SILU, bool OUTPLANES>
__global__ __launch_bounds__(256)
void gemm_bf16x3_kernel(const float* __restrict__ Ef32,
                        const __bf16* __restrict__ Ahi,
                        const __bf16* __restrict__ Alo,
                        const long long* __restrict__ tokens,
                        const float* __restrict__ Wf32,
                        const __bf16* __restrict__ Whi,
                        const __bf16* __restrict__ Wlo,
                        const float* __restrict__ bias,
                        float* __restrict__ Cf32,
                        __bf16* __restrict__ Chi,
                        __bf16* __restrict__ Clo,
                        int M, int N, int Kdim) {
    constexpr bool TDMA = (AMODE == 2);   // implies WMODE == 2

    __shared__ __align__(16) __bf16 AsH[2][BM * BK];
    __shared__ __align__(16) __bf16 AsL[2][BM * BK];
    __shared__ __align__(16) __bf16 BsH[2][BN * BK];
    __shared__ __align__(16) __bf16 BsL[2][BN * BK];

    const int tid  = threadIdx.x;
    const int lane = tid & 31;
    const int wv   = __builtin_amdgcn_readfirstlane(tid >> 5); // uniform wave id
    const int wm   = wv >> 2;          // 0..1 -> M sub-block of 32
    const int wn   = wv & 3;           // 0..3 -> N sub-block of 32
    const int n0   = blockIdx.x * BN;
    const int rsel = lane & 15;
    const int koff = (lane >> 4) << 3; // 0 or 8

    for (int mTile = blockIdx.y; mTile * BM < M; mTile += gridDim.y) {
        const int m0 = mTile * BM;

        v8f acc[2][2];
#pragma unroll
        for (int i = 0; i < 2; ++i)
#pragma unroll
            for (int j = 0; j < 2; ++j)
#pragma unroll
                for (int r = 0; r < 8; ++r) acc[i][j][r] = 0.0f;

        // ---- one k-step of fragment loads + 12 WMMAs from LDS buffer `buf`
        auto compute_step = [&](int buf) {
            v16bf ah[2], al[2], bh[2], bl[2];
#pragma unroll
            for (int i = 0; i < 2; ++i) {
                const int row = wm * 32 + i * 16 + rsel;
                ah[i] = load_frag16(&AsH[buf][row * BK + koff]);
                al[i] = load_frag16(&AsL[buf][row * BK + koff]);
            }
#pragma unroll
            for (int j = 0; j < 2; ++j) {
                const int col = wn * 32 + j * 16 + rsel;
                bh[j] = load_frag16(&BsH[buf][col * BK + koff]);
                bl[j] = load_frag16(&BsL[buf][col * BK + koff]);
            }
#pragma unroll
            for (int i = 0; i < 2; ++i)
#pragma unroll
                for (int j = 0; j < 2; ++j) {
                    acc[i][j] = __builtin_amdgcn_wmma_f32_16x16x32_bf16(
                        false, al[i], false, bh[j], (short)0, acc[i][j], false, false);
                    acc[i][j] = __builtin_amdgcn_wmma_f32_16x16x32_bf16(
                        false, ah[i], false, bl[j], (short)0, acc[i][j], false, false);
                    acc[i][j] = __builtin_amdgcn_wmma_f32_16x16x32_bf16(
                        false, ah[i], false, bh[j], (short)0, acc[i][j], false, false);
                }
        };

        if (TDMA) {
            // Waves 0..3 each own one tile DMA: A_hi, A_lo, B_hi, B_lo.
            // Descriptor group1 + LDS offsets are k-loop invariant; only the
            // global address (group0) changes per k-step.
            const __bf16* gbase = nullptr;
            unsigned      lds0 = 0, lds1 = 0;
            v8u           g1{};
            if (wv == 0) {
                gbase = Ahi + (size_t)m0 * Kdim;
                lds0  = lds_off_u32(&AsH[0][0]);
                lds1  = lds_off_u32(&AsH[1][0]);
                g1    = tdm_g1_2d_bf16(BK, BM, (unsigned)Kdim,
                                       (unsigned)(M - m0), (unsigned)Kdim);
            } else if (wv == 1) {
                gbase = Alo + (size_t)m0 * Kdim;
                lds0  = lds_off_u32(&AsL[0][0]);
                lds1  = lds_off_u32(&AsL[1][0]);
                g1    = tdm_g1_2d_bf16(BK, BM, (unsigned)Kdim,
                                       (unsigned)(M - m0), (unsigned)Kdim);
            } else if (wv == 2) {
                gbase = Whi + (size_t)n0 * Kdim;
                lds0  = lds_off_u32(&BsH[0][0]);
                lds1  = lds_off_u32(&BsH[1][0]);
                g1    = tdm_g1_2d_bf16(BK, BN, (unsigned)Kdim,
                                       (unsigned)(N - n0), (unsigned)Kdim);
            } else if (wv == 3) {
                gbase = Wlo + (size_t)n0 * Kdim;
                lds0  = lds_off_u32(&BsL[0][0]);
                lds1  = lds_off_u32(&BsL[1][0]);
                g1    = tdm_g1_2d_bf16(BK, BN, (unsigned)Kdim,
                                       (unsigned)(N - n0), (unsigned)Kdim);
            }
            const int T = Kdim / BK;
            if (wv < 4) tdm_issue(lds0, gbase, g1);
            for (int kt = 0; kt < T; ++kt) {
                const int cur = kt & 1;
                if (wv < 4) __builtin_amdgcn_s_wait_tensorcnt(0);
                __syncthreads();                       // buf[cur] visible
                if (wv < 4 && kt + 1 < T)              // DMA overlaps compute
                    tdm_issue(cur ? lds0 : lds1, gbase + (kt + 1) * BK, g1);
                compute_step(cur);
            }
            __syncthreads();  // done reading before next mTile's DMAs
        } else {
            // Token prefetch (AMODE==0): 2 rows x 3 window slots per thread,
            // hoisted fully out of the k-loop.
            const int mmA = tid >> 3;          // rows mmA and mmA+32
            const int e4  = (tid & 7) * 4;     // 4-elem k chunk
            long long tokA[KWIN][2];
            if (AMODE == 0) {
#pragma unroll
                for (int slot = 0; slot < KWIN; ++slot)
#pragma unroll
                    for (int r = 0; r < 2; ++r) {
                        const int m = m0 + mmA + r * 32;
                        const int s = m >> 1;          // BATCH == 2
                        const int b = m & 1;
                        const int t = s + slot - (KWIN - 1);
                        tokA[slot][r] =
                            (t >= 0) ? tokens[(long long)t * BATCH + b] : 0ll;
                    }
            }
            const bool nVec = ((N & 3) == 0);  // float4-safe rows (uniform)
            for (int kk = 0; kk < Kdim; kk += BK) {
                __syncthreads();
                if (AMODE == 0) {
                    // gather K-slice of concat(E[tok(s-2)],E[tok(s-1)],E[tok(s)])
                    const int slot = kk >> 10;            // BK | EMBED
                    const int eIn  = (kk & (EMBED - 1)) + e4;
#pragma unroll
                    for (int r = 0; r < 2; ++r) {
                        const long long tk =
                            (slot == 0) ? tokA[0][r]
                                        : ((slot == 1) ? tokA[1][r] : tokA[2][r]);
                        const float4 v =
                            *(const float4*)&Ef32[tk * (long long)EMBED + eIn];
                        v4bf hi, lo;
                        const float* vp = (const float*)&v;
#pragma unroll
                        for (int q = 0; q < 4; ++q) {
                            const __bf16 h = (__bf16)vp[q];
                            hi[q] = h;
                            lo[q] = (__bf16)(vp[q] - (float)h);
                        }
                        const int o = (mmA + r * 32) * BK + e4;
                        *(v4bf*)&AsH[0][o] = hi;
                        *(v4bf*)&AsL[0][o] = lo;
                    }
                } else {  // AMODE == 1: copy prepacked bf16 planes
                    const int mm = tid >> 2;            // 0..63
                    const int e8 = (tid & 3) * 8;       // 8-half chunks
                    const size_t g = (size_t)(m0 + mm) * Kdim + kk + e8;
                    *(v8bf*)&AsH[0][mm * BK + e8] = *(const v8bf*)&Ahi[g];
                    *(v8bf*)&AsL[0][mm * BK + e8] = *(const v8bf*)&Alo[g];
                }
                {   // WMODE == 0: stage f32 W transposed to [n][k], split.
                    // float4 along N (independent b128 loads), scalar fallback
                    // for unaligned/ragged rows.
                    const int n4    = (tid & 31) * 4;   // 0..124
                    const int kBase = tid >> 5;         // 0..7
#pragma unroll
                    for (int k = kBase; k < BK; k += 8) {
                        const int gn = n0 + n4;
                        float4 v = make_float4(0.f, 0.f, 0.f, 0.f);
                        float* vp = (float*)&v;
                        if (nVec && gn + 3 < N) {
                            v = *(const float4*)&Wf32[(long long)(kk + k) * N + gn];
                        } else {
#pragma unroll
                            for (int q = 0; q < 4; ++q)
                                if (gn + q < N)
                                    vp[q] = Wf32[(long long)(kk + k) * N + gn + q];
                        }
#pragma unroll
                        for (int q = 0; q < 4; ++q) {
                            const __bf16 h = (__bf16)vp[q];
                            BsH[0][(n4 + q) * BK + k] = h;
                            BsL[0][(n4 + q) * BK + k] = (__bf16)(vp[q] - (float)h);
                        }
                    }
                }
                __syncthreads();
                compute_step(0);
            }
        }

        // ---- epilogue: C frag layout VGPR r -> M=r+(lane>=16?8:0), N=lane&15
#pragma unroll
        for (int i = 0; i < 2; ++i)
#pragma unroll
            for (int j = 0; j < 2; ++j) {
                const int gn = n0 + wn * 32 + j * 16 + rsel;
                if (gn < N) {
                    const float bv    = bias[gn];
                    const int   mBase = m0 + wm * 32 + i * 16 + koff;
#pragma unroll
                    for (int r = 0; r < 8; ++r) {
                        float v = acc[i][j][r] + bv;
                        if (SILU) v = silu_f(v);
                        const size_t o = (size_t)(mBase + r) * N + gn;
                        if (OUTPLANES) {
                            const __bf16 hi = (__bf16)v;
                            Chi[o] = hi;
                            Clo[o] = (__bf16)(v - (float)hi);
                        } else {
                            Cf32[o] = v;
                        }
                    }
                }
            }
    }
}

// ---------------------------------------------------------------------------
// Prepack: W (f32, [K][N]) -> K-major bf16 hi/lo planes ([N][K]).
// 32x32 LDS tile transpose keeps both sides coalesced.
// ---------------------------------------------------------------------------
__global__ __launch_bounds__(256)
void prepack_split_kernel(const float* __restrict__ W,
                          __bf16* __restrict__ Whi, __bf16* __restrict__ Wlo,
                          int Kd, int N) {
    __shared__ float tile[32][33];
    const int k0 = blockIdx.y * 32;
    const int n0 = blockIdx.x * 32;
    const int tx = threadIdx.x & 31;
    const int ty = threadIdx.x >> 5;   // 0..7
#pragma unroll
    for (int r = ty; r < 32; r += 8) {
        const int n = n0 + tx;
        tile[r][tx] = (n < N) ? W[(long long)(k0 + r) * N + n] : 0.0f;
    }
    __syncthreads();
#pragma unroll
    for (int r = ty; r < 32; r += 8) {
        const int n = n0 + r;
        if (n < N) {
            const float v  = tile[tx][r];
            const __bf16 hi = (__bf16)v;
            const size_t o  = (size_t)n * Kd + (k0 + tx);
            Whi[o] = hi;
            Wlo[o] = (__bf16)(v - (float)hi);
        }
    }
}

extern "C" void kernel_launch(void* const* d_in, const int* in_sizes, int n_in,
                              void* d_out, int out_size, void* d_ws, size_t ws_size,
                              hipStream_t stream) {
    (void)in_sizes; (void)n_in; (void)out_size;

    const long long* tokens = (const long long*)d_in[0];  // int64 (SEQ,BATCH)
    const float* E    = (const float*)d_in[1];
    const float* W1   = (const float*)d_in[2];
    const float* b1   = (const float*)d_in[3];
    const float* W2   = (const float*)d_in[4];
    const float* b2   = (const float*)d_in[5];
    const float* Wout = (const float*)d_in[6];
    const float* bout = (const float*)d_in[7];
    float* out = (float*)d_out;

    // ---- workspace carving (256B aligned regions) ----
    size_t off = 0;
    auto carve = [&](size_t bytes) {
        void* p = (char*)d_ws + off;
        off += (bytes + 255) & ~(size_t)255;
        return p;
    };
    const size_t hB = (size_t)M_ROWS * EMBED * sizeof(__bf16);
    __bf16* h1h = (__bf16*)carve(hB);
    __bf16* h1l = (__bf16*)carve(hB);
    __bf16* h2h = (__bf16*)carve(hB);
    __bf16* h2l = (__bf16*)carve(hB);
    __bf16* w2h = (__bf16*)carve((size_t)EMBED * EMBED * sizeof(__bf16));
    __bf16* w2l = (__bf16*)carve((size_t)EMBED * EMBED * sizeof(__bf16));
    __bf16* woh = (__bf16*)carve((size_t)VOCAB * EMBED * sizeof(__bf16));
    __bf16* wol = (__bf16*)carve((size_t)VOCAB * EMBED * sizeof(__bf16));
    const bool full = (ws_size >= off);   // room for prepacked weight planes?

    dim3 blk(256);

    // h1 = silu(gather(E, tokens) @ W1 + b1)   M=4096 K=3072 N=1024
    gemm_bf16x3_kernel<0, 0, true, true>
        <<<dim3(EMBED / BN, M_ROWS / BM), blk, 0, stream>>>(
            E, nullptr, nullptr, tokens, W1, nullptr, nullptr, b1,
            nullptr, h1h, h1l, M_ROWS, EMBED, KWIN * EMBED);

    if (full) {
        prepack_split_kernel<<<dim3((EMBED + 31) / 32, EMBED / 32), blk, 0, stream>>>(
            W2, w2h, w2l, EMBED, EMBED);
        prepack_split_kernel<<<dim3((VOCAB + 31) / 32, EMBED / 32), blk, 0, stream>>>(
            Wout, woh, wol, EMBED, VOCAB);

        // h2 = silu(h1 @ W2 + b2)  — TDM double-buffered staging
        gemm_bf16x3_kernel<2, 2, true, true>
            <<<dim3(EMBED / BN, M_ROWS / BM), blk, 0, stream>>>(
                nullptr, h1h, h1l, nullptr, nullptr, w2h, w2l, b2,
                nullptr, h2h, h2l, M_ROWS, EMBED, EMBED);

        // logits = h2 @ Wout + bout; gridDim.y=4 -> Wout streamed only 4x
        // from HBM while h2 planes (16.8 MB) stay L2-resident.
        gemm_bf16x3_kernel<2, 2, false, false>
            <<<dim3((VOCAB + BN - 1) / BN, 4), blk, 0, stream>>>(
                nullptr, h2h, h2l, nullptr, nullptr, woh, wol, bout,
                out, nullptr, nullptr, M_ROWS, VOCAB, EMBED);
    } else {
        // Fallback: no room for weight planes -> convert W f32 while staging.
        gemm_bf16x3_kernel<1, 0, true, true>
            <<<dim3(EMBED / BN, M_ROWS / BM), blk, 0, stream>>>(
                nullptr, h1h, h1l, nullptr, W2, nullptr, nullptr, b2,
                nullptr, h2h, h2l, M_ROWS, EMBED, EMBED);
        gemm_bf16x3_kernel<1, 0, false, false>
            <<<dim3((VOCAB + BN - 1) / BN, 4), blk, 0, stream>>>(
                nullptr, h2h, h2l, nullptr, Wout, nullptr, nullptr, bout,
                out, nullptr, nullptr, M_ROWS, VOCAB, EMBED);
    }
}
